// RecurrentClassifier_59974923321592
// MI455X (gfx1250) — matrix operand order repaired and checked by reading, not compiled, not run on hardware
//
#include <hip/hip_runtime.h>

// ---------------- problem constants ----------------
#define BB   2048
#define DD   128
#define HH   256
#define G4   1024   // 4*H
#define NCC  10
#define TT   64
#define TICK 6
#define BT   16     // batch rows per block

typedef __attribute__((ext_vector_type(16))) __bf16 v16bf;
typedef __attribute__((ext_vector_type(8)))  float  v8f;

__device__ __forceinline__ float sigf(float x) { return 1.0f / (1.0f + __expf(-x)); }

// Prevent LICM/CSE from hoisting loop-invariant weight loads out of the
// tick/time loops (which caused massive scratch spilling): an opaque memory
// clobber forces re-load each iteration (the loads then stream from L2).
#define LOOP_FENCE() asm volatile("" ::: "memory")

// One wave computes a 16 x 128 slice (8 N-tiles of 16) of  D = A(16xK) * W^T + C
// A: LDS bf16 row-major [16][KTOT]   (CDNA5 A-fragment layout per ISA 7.12.2)
// W: global bf16 row-major [N][KTOT] (B-fragment: lane = column, contiguous K run)
// C: either LDS f32 [16][1024] (HAS_C) or a bias vector broadcast over rows
template <int KTOT, bool HAS_C>
__device__ __forceinline__ void wave_gemm(const __bf16* __restrict__ Asrc,
                                          const __bf16* __restrict__ Wg,
                                          const float* __restrict__ Cinit,
                                          const float* __restrict__ biasg,
                                          float* __restrict__ Dout,
                                          int wave, int lane) {
  const int nbase = wave * 128;
  const int nlo = lane & 15;
  const int hi  = lane >> 4;
  v8f acc[8];
  if (HAS_C) {
#pragma unroll
    for (int tt = 0; tt < 8; ++tt) {
      const int n = nbase + tt * 16 + nlo;
#pragma unroll
      for (int r = 0; r < 8; ++r) acc[tt][r] = Cinit[(r + hi * 8) * G4 + n];
    }
  } else {
#pragma unroll
    for (int tt = 0; tt < 8; ++tt) {
      const float bv = biasg[nbase + tt * 16 + nlo];
#pragma unroll
      for (int r = 0; r < 8; ++r) acc[tt][r] = bv;
    }
  }
#pragma unroll
  for (int kb = 0; kb < KTOT; kb += 32) {
    union { v16bf v; uint4 q[2]; } ua;
    const __bf16* pa = Asrc + nlo * KTOT + kb + hi * 8;
    ua.q[0] = *(const uint4*)(pa);
    ua.q[1] = *(const uint4*)(pa + 16);
#pragma unroll
    for (int tt = 0; tt < 8; ++tt) {
      union { v16bf v; uint4 q[2]; } ub;
      const __bf16* pb = Wg + (size_t)(nbase + tt * 16 + nlo) * KTOT + kb + hi * 8;
      ub.q[0] = *(const uint4*)(pb);
      ub.q[1] = *(const uint4*)(pb + 16);
      acc[tt] = __builtin_amdgcn_wmma_f32_16x16x32_bf16(
          false, ua.v, false, ub.v, (short)0, acc[tt], false, false);
    }
  }
#pragma unroll
  for (int tt = 0; tt < 8; ++tt) {
    const int n = nbase + tt * 16 + nlo;
#pragma unroll
    for (int r = 0; r < 8; ++r) Dout[(r + hi * 8) * G4 + n] = acc[tt][r];
  }
}

// Single-N-tile GEMM (wave 0 only): yh[16x16] = h(16x256) @ Wyh^T(256x16) + ybias
__device__ __forceinline__ void wave_gemm_yh(const __bf16* __restrict__ Asrc,
                                             const __bf16* __restrict__ Wg,
                                             const float* __restrict__ biasg,
                                             float* __restrict__ Dout, int lane) {
  const int nlo = lane & 15;
  const int hi  = lane >> 4;
  v8f acc;
  const float bv = biasg[nlo];
#pragma unroll
  for (int r = 0; r < 8; ++r) acc[r] = bv;
#pragma unroll
  for (int kb = 0; kb < HH; kb += 32) {
    union { v16bf v; uint4 q[2]; } ua, ub;
    const __bf16* pa = Asrc + nlo * HH + kb + hi * 8;
    ua.q[0] = *(const uint4*)(pa);
    ua.q[1] = *(const uint4*)(pa + 16);
    const __bf16* pb = Wg + (size_t)nlo * HH + kb + hi * 8;
    ub.q[0] = *(const uint4*)(pb);
    ub.q[1] = *(const uint4*)(pb + 16);
    acc = __builtin_amdgcn_wmma_f32_16x16x32_bf16(
        false, ua.v, false, ub.v, (short)0, acc, false, false);
  }
#pragma unroll
  for (int r = 0; r < 8; ++r) Dout[(r + hi * 8) * 16 + nlo] = acc[r];
}

// ---------------- prep: fp32 weights -> bf16, fused heads ----------------
__global__ void act_prep_kernel(const float* __restrict__ Wih,
                                const float* __restrict__ Whh,
                                const float* __restrict__ bih,
                                const float* __restrict__ bhh,
                                const float* __restrict__ Whalt,
                                const float* __restrict__ bhalt,
                                const float* __restrict__ Wdec,
                                const float* __restrict__ bdec,
                                unsigned short* __restrict__ WihBu,
                                unsigned short* __restrict__ WhhBu,
                                float* __restrict__ biasC,
                                unsigned short* __restrict__ WyhBu,
                                float* __restrict__ ybias) {
  __bf16* WihB = (__bf16*)WihBu;
  __bf16* WhhB = (__bf16*)WhhBu;
  __bf16* WyhB = (__bf16*)WyhBu;
  const int i = blockIdx.x * 256 + threadIdx.x;
  if (i < G4 * DD) WihB[i] = (__bf16)Wih[i];
  if (i < G4 * HH) WhhB[i] = (__bf16)Whh[i];
  if (i < G4)      biasC[i] = bih[i] + bhh[i];
  if (i < 16 * HH) {            // fused head matrix: rows 0..9 Wdec, row 10 Whalt
    const int n = i >> 8, j = i & (HH - 1);
    float v = 0.0f;
    if (n < NCC)       v = Wdec[n * HH + j];
    else if (n == NCC) v = Whalt[j];
    WyhB[i] = (__bf16)v;
  }
  if (i < 16) ybias[i] = (i < NCC) ? bdec[i] : ((i == NCC) ? bhalt[0] : 0.0f);
}

// ---------------- main persistent ACT-LSTM kernel ----------------
__global__ __launch_bounds__(256, 1)
void act_lstm_kernel(const float* __restrict__ x,
                     const unsigned short* __restrict__ WihU,
                     const unsigned short* __restrict__ WhhU,
                     const float* __restrict__ bias,
                     const unsigned short* __restrict__ WyhU,
                     const float* __restrict__ ybias,
                     float* __restrict__ Y,
                     float* __restrict__ P,
                     int* __restrict__ Nout,
                     float* __restrict__ Hh) {
  const __bf16* Wih = (const __bf16*)WihU;
  const __bf16* Whh = (const __bf16*)WhhU;
  const __bf16* Wyh = (const __bf16*)WyhU;

  __shared__ float  gxL[BT * G4];      // x@Wih^T + bias, per timestep      (64 KB)
  __shared__ float  gatesL[BT * G4];   // gate pre-activations; xi overlay  (64 KB)
  __shared__ __bf16 hbfL[BT * HH];     // h (bf16) GEMM operand             ( 8 KB)
  __shared__ float  cL[BT * HH];       // cell state fp32                   (16 KB)
  __shared__ float  yhL[BT * 16];      // fused head output [m][n] n<10=y, 10=halt
  __shared__ float  phL[BT];

  const int tid  = threadIdx.x;
  const int lane = tid & 31;
  const int wave = tid >> 5;
  const int b0   = blockIdx.x * BT;

  __bf16* xibf = (__bf16*)gatesL;      // overlay: xi bf16 only live while building gx

  for (int i = tid; i < BT * HH; i += 256) { hbfL[i] = (__bf16)0.0f; cL[i] = 0.0f; }

  // per-thread state
  float stR[BT], ctR[BT], h2R[BT];
  const int ym = tid / NCC, yc = tid - ym * NCC;
  const bool isY   = tid < BT * NCC;   // 160 threads own (m,c) output pairs
  const bool isACT = tid < BT;
  float ytR = 0.0f;
  float cumR = 0.0f, pmprevR = 0.0f, ptR = 0.0f, rtR = 0.0f;
  int foundR = 0, ntR = 0;

  __syncthreads();

  for (int t = 0; t < TT; ++t) {
    LOOP_FENCE();   // stop Wih/bias fragment hoisting across the time loop
    // ---- stage xi (strided gather from x[B,D,T]) as bf16 ----
#pragma unroll
    for (int r = 0; r < (BT * DD) / 256; ++r) {
      const int e = tid + r * 256;
      const int m = e >> 7, d = e & (DD - 1);
      xibf[e] = (__bf16)x[((size_t)(b0 + m) * DD + d) * TT + t];
    }
    __syncthreads();
    // ---- gx = xi @ Wih^T + (b_ih+b_hh) ----
    wave_gemm<DD, false>(xibf, Wih, nullptr, bias, gxL, wave, lane);
    __syncthreads();

#pragma unroll
    for (int m = 0; m < BT; ++m) { stR[m] = 0.0f; ctR[m] = 0.0f; }
    ytR = 0.0f; cumR = 0.0f; pmprevR = 0.0f; rtR = 0.0f; foundR = 0; ntR = 0;

    for (int k = 0; k < TICK; ++k) {
      LOOP_FENCE(); // stop Whh/gx fragment hoisting across the tick loop
      // ---- gates = gx + h @ Whh^T ----
      wave_gemm<HH, true>(hbfL, Whh, gxL, nullptr, gatesL, wave, lane);
      __syncthreads();

      // ---- LSTM cell elementwise: thread tid owns hidden unit j=tid ----
#pragma unroll
      for (int m = 0; m < BT; ++m) {
        const float gi = gatesL[m * G4 + tid];
        const float gf = gatesL[m * G4 + HH + tid];
        const float gg = gatesL[m * G4 + 2 * HH + tid];
        const float go = gatesL[m * G4 + 3 * HH + tid];
        const float c2 = sigf(gf) * cL[m * HH + tid] + sigf(gi) * tanhf(gg);
        const float h2 = sigf(go) * tanhf(c2);
        cL[m * HH + tid]   = c2;
        hbfL[m * HH + tid] = (__bf16)h2;
        h2R[m] = h2;
        Hh[(((size_t)(b0 + m) * HH + tid) * TICK + k) * TT + t] = h2;
      }
      __syncthreads();

      // ---- fused halt+decoder head via WMMA (wave 0 only) ----
      if (wave == 0) wave_gemm_yh(hbfL, Wyh, ybias, yhL, lane);
      __syncthreads();

      // ---- ACT bookkeeping per batch row ----
      if (isACT) {
        const float p = sigf(yhL[tid * 16 + NCC]);
        cumR += p;
        const float pm = (k == TICK - 1) ? 1.0f : fminf(1.0f, cumR);
        phL[tid] = pm - pmprevR;
        if (!foundR && pm >= 1.0f) {
          foundR = 1; ntR = k;
          rtR = (k == 0) ? 0.0f : (1.0f - pmprevR);
        }
        pmprevR = pm;
      }
      __syncthreads();

      // ---- ACT-weighted accumulation (registers) ----
#pragma unroll
      for (int m = 0; m < BT; ++m) {
        const float ph = phL[m];
        stR[m] += ph * h2R[m];
        ctR[m] += ph * cL[m * HH + tid];
      }
      if (isY) ytR += phL[ym] * yhL[ym * 16 + yc];
      __syncthreads();
    }

    // ---- timestep epilogue ----
    if (isY)  Y[((size_t)(b0 + ym) * NCC + yc) * TT + t] = ytR;
    if (isACT) {
      Nout[(size_t)(b0 + tid) * TT + t] = ntR;
      ptR += (float)ntR + rtR;
    }
#pragma unroll
    for (int m = 0; m < BT; ++m) {
      hbfL[m * HH + tid] = (__bf16)stR[m];   // carry h <- st
      cL[m * HH + tid]   = ctR[m];           // carry c <- ct
    }
    __syncthreads();
  }

  if (isACT) P[b0 + tid] = ptR;
}

extern "C" void kernel_launch(void* const* d_in, const int* in_sizes, int n_in,
                              void* d_out, int out_size, void* d_ws, size_t ws_size,
                              hipStream_t stream) {
  const float* x     = (const float*)d_in[0];
  const float* Wih   = (const float*)d_in[1];
  const float* Whh   = (const float*)d_in[2];
  const float* bih   = (const float*)d_in[3];
  const float* bhh   = (const float*)d_in[4];
  const float* Whalt = (const float*)d_in[5];
  const float* bhalt = (const float*)d_in[6];
  const float* Wdec  = (const float*)d_in[7];
  const float* bdec  = (const float*)d_in[8];

  // workspace layout (~800 KB): bf16 weights + fused bias + fused head
  char* ws = (char*)d_ws;
  unsigned short* WihB  = (unsigned short*)ws;                            // 1024*128*2
  unsigned short* WhhB  = (unsigned short*)(ws + 262144);                 // 1024*256*2
  float*          biasC = (float*)(ws + 262144 + 524288);                 // 1024*4
  unsigned short* WyhB  = (unsigned short*)(ws + 262144 + 524288 + 4096); // 16*256*2
  float*          ybias = (float*)(ws + 262144 + 524288 + 4096 + 8192);   // 16*4

  // outputs concatenated flat: Y [B,NC,T] | P [B] | N [B,T] int32 | Hh [B,H,TICK,T]
  float* out  = (float*)d_out;
  float* Y    = out;
  float* P    = out + (size_t)BB * NCC * TT;
  int*   Nout = (int*)(out + (size_t)BB * NCC * TT + BB);
  float* Hh   = out + (size_t)BB * NCC * TT + BB + (size_t)BB * TT;

  act_prep_kernel<<<(G4 * HH) / 256, 256, 0, stream>>>(
      Wih, Whh, bih, bhh, Whalt, bhalt, Wdec, bdec, WihB, WhhB, biasC, WyhB, ybias);
  act_lstm_kernel<<<BB / BT, 256, 0, stream>>>(x, WihB, WhhB, biasC, WyhB, ybias,
                                               Y, P, Nout, Hh);
}
